// SpiralConv_60344290508906
// MI455X (gfx1250) — matrix operand order repaired and checked by reading, not compiled, not run on hardware
//
#include <hip/hip_runtime.h>

typedef __attribute__((ext_vector_type(16))) _Float16 v16h;
typedef __attribute__((ext_vector_type(8)))  _Float16 v8h;
typedef __attribute__((ext_vector_type(8)))  float    v8f;
typedef __attribute__((ext_vector_type(4)))  float    v4f;

#define N_NODES_C 163842
#define SEQ_LEN_C 9
#define IN_CH_C   32
#define OUT_CH_C  64
#define K_TOT_C   (SEQ_LEN_C * IN_CH_C)   // 288
#define LDS_STRIDE_H 296                  // 288 + 8 halfword pad -> 592B row stride (bank-friendly)

__global__ __launch_bounds__(256)
void spiralconv_wmma_f16(const float* __restrict__ x,
                         const int*   __restrict__ idx,
                         const float* __restrict__ W,
                         const float* __restrict__ bias,
                         float* __restrict__ out,
                         int n_tiles)
{
    __shared__ _Float16 ldsW[OUT_CH_C * LDS_STRIDE_H];   // 37,888 B

    // Stage W (f32 -> f16) into LDS once per workgroup. W is [64][288] row-major.
    for (int i = threadIdx.x; i < OUT_CH_C * K_TOT_C; i += blockDim.x) {
        int o = i / K_TOT_C;
        int k = i - o * K_TOT_C;
        ldsW[o * LDS_STRIDE_H + k] = (_Float16)W[i];
    }
    __syncthreads();

    const int lane = threadIdx.x & 31;
    const int wave = threadIdx.x >> 5;
    const int tile = blockIdx.x * (blockDim.x >> 5) + wave;
    if (tile >= n_tiles) return;

    const int half = lane >> 4;   // selects K-subblock per the 16-bit A/B VGPR layouts
    const int n    = lane & 15;   // column within a 16-wide B/C block; also A row
    const int row  = lane & 15;

    int node        = tile * 16 + row;
    int nodeClamped = node < N_NODES_C ? node : (N_NODES_C - 1);

    v8f c0 = {}, c1 = {}, c2 = {}, c3 = {};   // 16x64 f32 accumulators

    for (int s = 0; s < SEQ_LEN_C; ++s) {
        // ---- A: gather one neighbor row; this lane covers channels
        //      [half*8, half*8+8) and [16+half*8, 16+half*8+8)  (ISA 16-bit A layout)
        int g = idx[nodeClamped * SEQ_LEN_C + s];
        const float* xr = x + (size_t)g * IN_CH_C;
        const v4f* p = (const v4f*)(xr + half * 8);
        const v4f* q = (const v4f*)(xr + 16 + half * 8);
        v4f p0 = p[0], p1 = p[1];
        v4f q0 = q[0], q1 = q[1];

        v16h a;
        a[0]  = (_Float16)p0[0]; a[1]  = (_Float16)p0[1];
        a[2]  = (_Float16)p0[2]; a[3]  = (_Float16)p0[3];
        a[4]  = (_Float16)p1[0]; a[5]  = (_Float16)p1[1];
        a[6]  = (_Float16)p1[2]; a[7]  = (_Float16)p1[3];
        a[8]  = (_Float16)q0[0]; a[9]  = (_Float16)q0[1];
        a[10] = (_Float16)q0[2]; a[11] = (_Float16)q0[3];
        a[12] = (_Float16)q1[0]; a[13] = (_Float16)q1[1];
        a[14] = (_Float16)q1[2]; a[15] = (_Float16)q1[3];

        // ---- B: W^T slices from LDS. Lane needs W[o][kbase .. kbase+7] and W[o][kbase+16 .. kbase+23],
        //      o = nb*16 + n, kbase = s*32 + half*8  (ISA 16-bit B layout)
        int kbase = s * IN_CH_C + half * 8;
        #pragma unroll
        for (int nb = 0; nb < 4; ++nb) {
            const _Float16* lw = &ldsW[(nb * 16 + n) * LDS_STRIDE_H + kbase];
            v8h b0 = *(const v8h*)lw;
            v8h b1 = *(const v8h*)(lw + 16);
            v16h bmat = __builtin_shufflevector(b0, b1,
                        0,1,2,3,4,5,6,7,8,9,10,11,12,13,14,15);
            v8f& c = (nb == 0) ? c0 : (nb == 1) ? c1 : (nb == 2) ? c2 : c3;
            c = __builtin_amdgcn_wmma_f32_16x16x32_f16(
                    /*neg_a=*/false, a, /*neg_b=*/false, bmat,
                    /*c_mod=*/(short)0, c, /*reuse_a=*/false, /*reuse_b=*/false);
        }
    }

    // ---- bias + store: C layout -> lane holds (M = v + half*8, N = n) for VGPR v
    float b0v = bias[n];
    float b1v = bias[16 + n];
    float b2v = bias[32 + n];
    float b3v = bias[48 + n];

    #pragma unroll
    for (int v = 0; v < 8; ++v) {
        int m  = v + half * 8;
        int nd = tile * 16 + m;
        if (nd < N_NODES_C) {
            float* orow = out + (size_t)nd * OUT_CH_C + n;
            orow[0]  = c0[v] + b0v;
            orow[16] = c1[v] + b1v;
            orow[32] = c2[v] + b2v;
            orow[48] = c3[v] + b3v;
        }
    }
}

extern "C" void kernel_launch(void* const* d_in, const int* in_sizes, int n_in,
                              void* d_out, int out_size, void* d_ws, size_t ws_size,
                              hipStream_t stream) {
    const float* x   = (const float*)d_in[0];
    const int*   idx = (const int*)d_in[1];
    const float* W   = (const float*)d_in[2];
    const float* b   = (const float*)d_in[3];
    float* out = (float*)d_out;

    int n_tiles = (N_NODES_C + 15) / 16;              // 10241 tiles of 16 nodes
    int waves_per_block = 8;                          // 256 threads, wave32
    int blocks = (n_tiles + waves_per_block - 1) / waves_per_block;

    spiralconv_wmma_f16<<<blocks, 256, 0, stream>>>(x, idx, W, b, out, n_tiles);
}
